// MultiHeadSelfAttention_34308198761156
// MI455X (gfx1250) — matrix operand order repaired and checked by reading, not compiled, not run on hardware
//
#include <hip/hip_runtime.h>
#include <hip/hip_bf16.h>
#include <stdint.h>

// ---------------------------------------------------------------------------
// Types
// ---------------------------------------------------------------------------
typedef __attribute__((ext_vector_type(16))) __bf16       v16bf;
typedef __attribute__((ext_vector_type(8)))  float        v8f;
typedef __attribute__((ext_vector_type(4)))  unsigned int u32x4;
typedef __attribute__((ext_vector_type(8)))  int          i32x8;
typedef __attribute__((ext_vector_type(4)))  int          i32x4;

union BFrag { unsigned int u[8]; v16bf v; };

#define B_    2
#define S_    2048
#define D_    1024
#define H_    16
#define HD_   64
#define MTOT  (B_ * S_)   // 4096

__device__ __forceinline__ unsigned short f2bf(float f) {
  unsigned int u = __float_as_uint(f);
  u += 0x7FFFu + ((u >> 16) & 1u);      // round-to-nearest-even
  return (unsigned short)(u >> 16);
}

// ---------------------------------------------------------------------------
// DPP16 cross-lane reductions over each 16-lane half (wave32), pure VALU.
// xor1 = quad_perm(1,0,3,2)=0xB1, xor2 = quad_perm(2,3,0,1)=0x4E,
// then ROW_HALF_MIRROR (0x141) and ROW_MIRROR (0x140) finish the 16 lanes.
// ---------------------------------------------------------------------------
template <int CTRL>
__device__ __forceinline__ float dppf(float x) {
  return __int_as_float(__builtin_amdgcn_update_dpp(
      0, __float_as_int(x), CTRL, 0xF, 0xF, true));
}
__device__ __forceinline__ float rowmax16(float x) {
  x = fmaxf(x, dppf<0xB1>(x));
  x = fmaxf(x, dppf<0x4E>(x));
  x = fmaxf(x, dppf<0x141>(x));
  x = fmaxf(x, dppf<0x140>(x));
  return x;
}
__device__ __forceinline__ float rowsum16(float x) {
  x += dppf<0xB1>(x);
  x += dppf<0x4E>(x);
  x += dppf<0x141>(x);
  x += dppf<0x140>(x);
  return x;
}

// ---------------------------------------------------------------------------
// TDM: generic 2D tile load (global -> LDS) with LDS row padding.
// 6-arg builtin form: (u32x4 g0, i32x8 g1, i32x4, i32x4, i32x8, i32 cpol)
// ---------------------------------------------------------------------------
#if __has_builtin(__builtin_amdgcn_tensor_load_to_lds)
#define HAVE_TDM 1
__device__ __forceinline__ void tdm_load_2d(
    unsigned int lds_off, const void* gptr, unsigned int dsz_code,
    unsigned int tile_d0, unsigned int tile_d1,
    unsigned int tensor_d0, unsigned int tensor_d1,
    unsigned int stride_d0, unsigned int pad_int, unsigned int pad_amt) {
  unsigned long long ga = (unsigned long long)(uintptr_t)gptr;
  u32x4 g0;
  g0[0] = 1u;                                              // count=1, user D#
  g0[1] = lds_off;                                         // LDS byte address
  g0[2] = (unsigned int)ga;                                // global_addr lo
  g0[3] = ((unsigned int)(ga >> 32) & 0x01FFFFFFu) | (2u << 30); // hi + type=2
  i32x8 g1;
  g1[0] = (int)((dsz_code << 16) | (1u << 20) |            // data_size, pad_en
                (pad_int << 22) | (pad_amt << 25));
  g1[1] = (int)((tensor_d0 & 0xFFFFu) << 16);
  g1[2] = (int)((tensor_d0 >> 16) | ((tensor_d1 & 0xFFFFu) << 16));
  g1[3] = (int)(((tensor_d1 >> 16) & 0xFFFFu) | (tile_d0 << 16));
  g1[4] = (int)tile_d1;                                    // tile_dim2 = 0
  g1[5] = (int)stride_d0;                                  // 48-bit stride lo
  g1[6] = 0;
  g1[7] = 0;
  i32x4 gz4 = {0, 0, 0, 0};
  i32x8 gz8 = {0, 0, 0, 0, 0, 0, 0, 0};
  __builtin_amdgcn_tensor_load_to_lds(g0, g1, gz4, gz4, gz8, 0);
}
#else
#define HAVE_TDM 0
#endif

// ---------------------------------------------------------------------------
// Kernel 1a: plain f32 -> bf16
// ---------------------------------------------------------------------------
__global__ void cvt_f32_bf16(const float* __restrict__ src,
                             unsigned short* __restrict__ dst, int n) {
  int i = blockIdx.x * blockDim.x + threadIdx.x;
  int stride = gridDim.x * blockDim.x;
  for (; i < n; i += stride) dst[i] = f2bf(src[i]);
}
// Kernel 1b: weights f32 [k][n] -> bf16 pair-packed [(k/2)][n][k%2]
// so a WMMA B-fragment reads contraction pairs as single dwords.
__global__ void cvt_pack_w(const float* __restrict__ src,
                           unsigned short* __restrict__ dst, int n) {
  int i = blockIdx.x * blockDim.x + threadIdx.x;
  int stride = gridDim.x * blockDim.x;
  for (; i < n; i += stride) {
    int k = i >> 10, c = i & 1023;
    dst[(((size_t)(k >> 1) * D_) + c) * 2 + (k & 1)] = f2bf(src[i]);
  }
}

// ---------------------------------------------------------------------------
// Kernel 2/4: bf16 WMMA GEMM  C[4096x1024] = A * W + b
// MODE 0: bf16 out, [B][H][S][HD] layout, scaled   (Q / K projection)
// MODE 2: bf16 out, [B][H][HD][S] layout (V^T)     (V projection)
// MODE 1: f32 out, row-major [4096][1024]          (output projection)
// Block: 256 threads (8 waves). Tile: 128(M) x 64(N), K-step 32.
// A and pair-packed B tiles staged by TDM, double-buffered.
// ---------------------------------------------------------------------------
#define MT     128
#define NT     64
#define KSTEP  32
#define A_PAD  40   // LDS A row stride (ushorts): 32 data + 8 pad (TDM pad)
#define BP_PAD 66   // LDS B row stride (uints):   64 data + 2 pad (TDM pad)

template <int MODE>
__global__ __launch_bounds__(256) void gemm_bf16(
    const unsigned short* __restrict__ A,   // [4096][1024] bf16 row-major
    const unsigned int* __restrict__ Wp,    // [512][1024] k-pair-packed dwords
    const float* __restrict__ bias,         // [1024]
    unsigned short* __restrict__ outh,      // MODE 0/2 target
    float* __restrict__ outf,               // MODE 1 target
    float scale) {
  __shared__ unsigned short At[2 * MT * A_PAD];
  __shared__ unsigned int   Bp[2 * 16 * BP_PAD];

  const int tid   = threadIdx.x;
  const int lane  = tid & 31;
  const int wv    = tid >> 5;
  const int mbase = blockIdx.y * MT;
  const int nbase = blockIdx.x * NT;

  const v8f vzero = {0.f, 0.f, 0.f, 0.f, 0.f, 0.f, 0.f, 0.f};
  v8f acc[4] = {vzero, vzero, vzero, vzero};

  int buf = 0;
#if HAVE_TDM
  if (wv == 0) {  // prime buffer 0 (k0 = 0)
    tdm_load_2d((unsigned int)(uintptr_t)&At[0], A + (size_t)mbase * D_,
                1, KSTEP, MT, D_, MTOT, D_, /*pad*/3, 3);
    tdm_load_2d((unsigned int)(uintptr_t)&Bp[0], Wp + nbase,
                2, 64, 16, D_, D_ / 2, D_, /*pad*/5, 1);
  }
#endif

  for (int k0 = 0; k0 < D_; k0 += KSTEP) {
    __syncthreads();                       // prior compute done on buf^1
#if HAVE_TDM
    if (wv == 0) {
      if (k0 + KSTEP < D_) {               // prefetch next tile, then wait
        int nb = buf ^ 1, k1 = k0 + KSTEP; // for current tile (in-order TDM)
        tdm_load_2d((unsigned int)(uintptr_t)&At[nb * MT * A_PAD],
                    A + (size_t)mbase * D_ + k1,
                    1, KSTEP, MT, D_, MTOT, D_, 3, 3);
        tdm_load_2d((unsigned int)(uintptr_t)&Bp[nb * 16 * BP_PAD],
                    Wp + (size_t)(k1 >> 1) * D_ + nbase,
                    2, 64, 16, D_, D_ / 2, D_, 5, 1);
        __builtin_amdgcn_s_wait_tensorcnt(2);
      } else {
        __builtin_amdgcn_s_wait_tensorcnt(0);
      }
    }
#else
#pragma unroll
    for (int i = 0; i < 4; i++) {          // A tile 128x32
      int lin = tid + i * 256;
      int row = lin >> 3, c4 = (lin & 7) << 2;
      *(uint2*)&At[buf * MT * A_PAD + row * A_PAD + c4] =
          *(const uint2*)(A + (size_t)(mbase + row) * D_ + k0 + c4);
    }
#pragma unroll
    for (int i = 0; i < 2; i++) {          // B tile 16x64 dwords
      int lin = tid + i * 256;
      int row = lin >> 5, c2 = (lin & 31) << 1;
      *(uint2*)&Bp[buf * 16 * BP_PAD + row * BP_PAD + c2] =
          *(const uint2*)(Wp + (size_t)((k0 >> 1) + row) * D_ + nbase + c2);
    }
#endif
    __syncthreads();                       // tile in buf ready

    const unsigned short* AtL = &At[buf * MT * A_PAD];
    const unsigned int*   BpL = &Bp[buf * 16 * BP_PAD];

    BFrag af;
    {
      int m  = wv * 16 + (lane & 15);
      int kh = (lane < 16) ? 0 : 8;
#pragma unroll
      for (int v = 0; v < 8; v++) {
        int k = ((v & 4) ? 16 : 0) + kh + ((v & 3) << 1);
        af.u[v] = *(const unsigned int*)&AtL[m * A_PAD + k];
      }
    }
    int kb2 = (lane < 16) ? 0 : 4;         // pair-row base per lane half
#pragma unroll
    for (int nt = 0; nt < 4; nt++) {
      BFrag bf;
      int n = nt * 16 + (lane & 15);
#pragma unroll
      for (int v = 0; v < 8; v++) bf.u[v] = BpL[(kb2 + (v & 3) + ((v & 4) ? 8 : 0)) * BP_PAD + n];
      acc[nt] = __builtin_amdgcn_wmma_f32_16x16x32_bf16(
          false, af.v, false, bf.v, (short)0, acc[nt], false, false);
    }
    buf ^= 1;
  }

  // --- epilogue ---
#pragma unroll
  for (int nt = 0; nt < 4; nt++) {
    int n = nbase + nt * 16 + (lane & 15);
    float bn = bias[n];
#pragma unroll
    for (int r = 0; r < 8; r++) {
      int m = mbase + wv * 16 + r + ((lane < 16) ? 0 : 8);
      float val = (acc[nt][r] + bn) * scale;
      int b = m >> 11, s = m & (S_ - 1);
      int h = n >> 6, hd = n & (HD_ - 1);
      if (MODE == 0) {
        outh[(((size_t)(b * H_ + h) * S_) + s) * HD_ + hd] = f2bf(val);
      } else if (MODE == 2) {              // V stored transposed per head
        outh[(((size_t)(b * H_ + h) * HD_) + hd) * S_ + s] = f2bf(val);
      } else {
        outf[(size_t)m * D_ + n] = val;
      }
    }
  }
}

// ---------------------------------------------------------------------------
// Kernel 3: flash attention.  grid = (S/128, B*H), block = 256 (8 waves).
// K tiles [64 keys][64 hd] and V^T tiles [64 hd][64 keys] stream through LDS
// via double-buffered TDM; Q pre-scaled by 1/sqrt(HD).
// ---------------------------------------------------------------------------
#define QT     128
#define KTILE  64
#define KV_PAD 72   // 64 data + 8 pad (TDM pad: 4 DWORDs per 32-DWORD row)

__global__ __launch_bounds__(256) void attn_kernel(
    const unsigned short* __restrict__ Q,   // [BH][S][HD] bf16 (scaled)
    const unsigned short* __restrict__ K,   // [BH][S][HD] bf16
    const unsigned short* __restrict__ Vt,  // [BH][HD][S] bf16 (transposed)
    unsigned short* __restrict__ Aout) {    // [B*S][D] bf16
  __shared__ unsigned short Kt[2 * KTILE * KV_PAD];
  __shared__ unsigned short Vs[2 * HD_ * KV_PAD];
  __shared__ unsigned short Pst[8 * 16 * KV_PAD];

  const int tid   = threadIdx.x;
  const int lane  = tid & 31;
  const int wv    = tid >> 5;
  const int bh    = blockIdx.y;
  const int qbase = blockIdx.x * QT;

  const unsigned short* Qp  = Q  + (size_t)bh * S_ * HD_;
  const unsigned short* Kp  = K  + (size_t)bh * S_ * HD_;
  const unsigned short* Vpt = Vt + (size_t)bh * HD_ * S_;

  // ---- this wave's Q fragments (A layout), hd chunks c = 0,1 ----
  BFrag qa[2];
  {
    int srow = qbase + wv * 16 + (lane & 15);
    int kh   = (lane < 16) ? 0 : 8;
#pragma unroll
    for (int c = 0; c < 2; c++)
#pragma unroll
      for (int v = 0; v < 8; v++) {
        int k = c * 32 + ((v & 4) ? 16 : 0) + kh + ((v & 3) << 1);
        qa[c].u[v] = *(const unsigned int*)(Qp + (size_t)srow * HD_ + k);
      }
  }

  const v8f vzero = {0.f, 0.f, 0.f, 0.f, 0.f, 0.f, 0.f, 0.f};
  v8f oacc[4] = {vzero, vzero, vzero, vzero};
  float mrow[8], lrow[8];
#pragma unroll
  for (int r = 0; r < 8; r++) { mrow[r] = -3.0e38f; lrow[r] = 0.f; }

  int buf = 0;
#if HAVE_TDM
  if (wv == 0) {  // prime buffer 0 (j = 0)
    tdm_load_2d((unsigned int)(uintptr_t)&Kt[0], Kp,
                1, 64, 64, 64, S_, 64, 4, 3);
    tdm_load_2d((unsigned int)(uintptr_t)&Vs[0], Vpt,
                1, 64, 64, S_, HD_, S_, 4, 3);
  }
#endif

  for (int j = 0; j < S_; j += KTILE) {
    __syncthreads();                       // prior compute done on buf^1
#if HAVE_TDM
    if (wv == 0) {
      if (j + KTILE < S_) {
        int nb = buf ^ 1, j1 = j + KTILE;
        tdm_load_2d((unsigned int)(uintptr_t)&Kt[nb * KTILE * KV_PAD],
                    Kp + (size_t)j1 * HD_, 1, 64, 64, 64, S_, 64, 4, 3);
        tdm_load_2d((unsigned int)(uintptr_t)&Vs[nb * HD_ * KV_PAD],
                    Vpt + j1, 1, 64, 64, S_, HD_, S_, 4, 3);
        __builtin_amdgcn_s_wait_tensorcnt(2);
      } else {
        __builtin_amdgcn_s_wait_tensorcnt(0);
      }
    }
#else
#pragma unroll
    for (int i = 0; i < 4; i++) {
      int lin = tid + i * 256;
      int row = lin >> 4, c4 = (lin & 15) << 2;
      *(uint2*)&Kt[buf * KTILE * KV_PAD + row * KV_PAD + c4] =
          *(const uint2*)(Kp + (size_t)(j + row) * HD_ + c4);
      *(uint2*)&Vs[buf * HD_ * KV_PAD + row * KV_PAD + c4] =
          *(const uint2*)(Vpt + (size_t)row * S_ + j + c4);
    }
#endif
    __syncthreads();                       // tiles in buf ready

    const unsigned short* KtL = &Kt[buf * KTILE * KV_PAD];
    const unsigned short* VsL = &Vs[buf * HD_ * KV_PAD];

    // ---- scores: S = Q * K^T (16 queries x 64 keys) ----
    v8f sacc[4];
#pragma unroll
    for (int nt = 0; nt < 4; nt++) {
      BFrag kb0, kb1;
      int nk = nt * 16 + (lane & 15);      // key (B-matrix column)
      int kb = (lane < 16) ? 0 : 8;        // contraction (hd) half
#pragma unroll
      for (int v = 0; v < 8; v++) {
        kb0.u[v] = *(const unsigned int*)&KtL[nk * KV_PAD + kb + ((v & 4) ? 16 : 0) + 2 * (v & 3)];
        kb1.u[v] = *(const unsigned int*)&KtL[nk * KV_PAD + 32 + kb + ((v & 4) ? 16 : 0) + 2 * (v & 3)];
      }
      v8f s0 = vzero;
      s0 = __builtin_amdgcn_wmma_f32_16x16x32_bf16(
          false, qa[0].v, false, kb0.v, (short)0, s0, false, false);
      s0 = __builtin_amdgcn_wmma_f32_16x16x32_bf16(
          false, qa[1].v, false, kb1.v, (short)0, s0, false, false);
      sacc[nt] = s0;
    }

    // ---- online softmax: DPP16 row reductions (no LDS traffic) ----
#pragma unroll
    for (int r = 0; r < 8; r++) {
      float mx = rowmax16(fmaxf(fmaxf(sacc[0][r], sacc[1][r]),
                                fmaxf(sacc[2][r], sacc[3][r])));
      float mnew  = fmaxf(mrow[r], mx);
      float alpha = __expf(mrow[r] - mnew);
      mrow[r] = mnew;
      float ls = 0.f;
#pragma unroll
      for (int nt = 0; nt < 4; nt++) {
        float p = __expf(sacc[nt][r] - mnew);
        sacc[nt][r] = p;
        ls += p;
      }
      lrow[r] = lrow[r] * alpha + rowsum16(ls);
#pragma unroll
      for (int nt = 0; nt < 4; nt++) oacc[nt][r] *= alpha;
    }

    // ---- stage P (C layout -> LDS) to re-fragment as A matrix ----
    unsigned short* Pw = &Pst[wv * 16 * KV_PAD];
#pragma unroll
    for (int nt = 0; nt < 4; nt++) {
      int col = nt * 16 + (lane & 15);
#pragma unroll
      for (int r = 0; r < 8; r++) {
        int row = r + ((lane < 16) ? 0 : 8);
        Pw[row * KV_PAD + col] = f2bf(sacc[nt][r]);
      }
    }
    asm volatile("" ::: "memory");         // same-wave LDS is in-order;
    __builtin_amdgcn_wave_barrier();       // stop compiler reordering only

    // ---- O += P * V (contraction over 64 keys; V^T gives dword pairs) ----
#pragma unroll
    for (int c = 0; c < 2; c++) {
      BFrag pa;
      int row = lane & 15;
      int kh  = (lane < 16) ? 0 : 8;
#pragma unroll
      for (int v = 0; v < 8; v++) {
        int key = c * 32 + ((v & 4) ? 16 : 0) + kh + ((v & 3) << 1);
        pa.u[v] = *(const unsigned int*)&Pw[row * KV_PAD + key];
      }
      int kb = c * 32 + ((lane < 16) ? 0 : 8);
#pragma unroll
      for (int nt = 0; nt < 4; nt++) {
        BFrag vb;
        int hd = nt * 16 + (lane & 15);
#pragma unroll
        for (int v = 0; v < 8; v++)
          vb.u[v] = *(const unsigned int*)&VsL[hd * KV_PAD + kb + ((v & 4) ? 16 : 0) + 2 * (v & 3)];
        oacc[nt] = __builtin_amdgcn_wmma_f32_16x16x32_bf16(
            false, pa.v, false, vb.v, (short)0, oacc[nt], false, false);
      }
    }
    buf ^= 1;
  }

  // ---- normalize, write attn output (bf16, head-concat layout) ----
  const int b = bh >> 4;
  const int h = bh & (H_ - 1);
#pragma unroll
  for (int r = 0; r < 8; r++) {
    float inv  = 1.0f / lrow[r];
    int   srow = qbase + wv * 16 + r + ((lane < 16) ? 0 : 8);
#pragma unroll
    for (int nt = 0; nt < 4; nt++) {
      int hd = nt * 16 + (lane & 15);
      Aout[(size_t)(b * S_ + srow) * D_ + h * HD_ + hd] =
          f2bf(oacc[nt][r] * inv);
    }
  }
}

// ---------------------------------------------------------------------------
// Host launcher
// ---------------------------------------------------------------------------
extern "C" void kernel_launch(void* const* d_in, const int* in_sizes, int n_in,
                              void* d_out, int out_size, void* d_ws,
                              size_t ws_size, hipStream_t stream) {
  const float* x  = (const float*)d_in[0];
  const float* Wq = (const float*)d_in[1];
  const float* bq = (const float*)d_in[2];
  const float* Wk = (const float*)d_in[3];
  const float* bk = (const float*)d_in[4];
  const float* Wv = (const float*)d_in[5];
  const float* bv = (const float*)d_in[6];
  const float* Wo = (const float*)d_in[7];
  const float* bo = (const float*)d_in[8];
  float* out = (float*)d_out;

  char* ws = (char*)d_ws;
  const size_t MB = 1024 * 1024;
  unsigned short* xh  = (unsigned short*)(ws + 0 * MB);   // 8 MiB
  unsigned short* Wqh = (unsigned short*)(ws + 8 * MB);   // 2 MiB each, packed
  unsigned short* Wkh = (unsigned short*)(ws + 10 * MB);
  unsigned short* Wvh = (unsigned short*)(ws + 12 * MB);
  unsigned short* Woh = (unsigned short*)(ws + 14 * MB);
  unsigned short* Qh  = (unsigned short*)(ws + 16 * MB);  // 8 MiB each
  unsigned short* Kh  = (unsigned short*)(ws + 24 * MB);
  unsigned short* Vth = (unsigned short*)(ws + 32 * MB);  // V^T per head
  unsigned short* Ah  = (unsigned short*)(ws + 40 * MB);

  cvt_f32_bf16<<<2048, 256, 0, stream>>>(x, xh, MTOT * D_);
  cvt_pack_w<<<1024, 256, 0, stream>>>(Wq, Wqh, D_ * D_);
  cvt_pack_w<<<1024, 256, 0, stream>>>(Wk, Wkh, D_ * D_);
  cvt_pack_w<<<1024, 256, 0, stream>>>(Wv, Wvh, D_ * D_);
  cvt_pack_w<<<1024, 256, 0, stream>>>(Wo, Woh, D_ * D_);

  dim3 gg(D_ / NT, MTOT / MT);  // (16, 32)
  const float qscale = 0.125f;  // 1/sqrt(HD)
  gemm_bf16<0><<<gg, 256, 0, stream>>>(xh, (const unsigned int*)Wqh, bq, Qh,
                                       nullptr, qscale);
  gemm_bf16<0><<<gg, 256, 0, stream>>>(xh, (const unsigned int*)Wkh, bk, Kh,
                                       nullptr, 1.0f);
  gemm_bf16<2><<<gg, 256, 0, stream>>>(xh, (const unsigned int*)Wvh, bv, Vth,
                                       nullptr, 1.0f);

  attn_kernel<<<dim3(S_ / QT, B_ * H_), 256, 0, stream>>>(Qh, Kh, Vth, Ah);

  gemm_bf16<1><<<gg, 256, 0, stream>>>(Ah, (const unsigned int*)Woh, bo,
                                       nullptr, out, 1.0f);
}
